// Attention_27925877358942
// MI455X (gfx1250) — matrix operand order repaired and checked by reading, not compiled, not run on hardware
//
#include <hip/hip_runtime.h>
#include <math.h>

// ---------------- CDNA5 WMMA types ----------------
typedef __attribute__((ext_vector_type(16))) __bf16 v16bf;
typedef __attribute__((ext_vector_type(8)))  float  v8f;

struct U4 { unsigned x, y, z, w; };
union V16 { v16bf v; unsigned short u[16]; U4 q[2]; };
union P8  { U4 q; unsigned short u[8]; };

// ---------------- Problem constants ----------------
#define NN   4096      // sequence length
#define DM   1536      // model dim
#define HH   8         // heads
#define DD   64        // head dim
#define HD   512       // HH*DD
#define FF   192       // rel-pos features
#define LL   (2*NN-1)  // 8191
#define QSCALE 0.125f  // 64^-0.5

// ---------------- helpers ----------------
static __device__ inline unsigned short f2bf(float f) {
    unsigned u = __float_as_uint(f);
    u += 0x7fffu + ((u >> 16) & 1u);           // round-to-nearest-even
    return (unsigned short)(u >> 16);
}

static __device__ inline v8f vzero8() {
    v8f z;
#pragma unroll
    for (int i = 0; i < 8; ++i) z[i] = 0.0f;
    return z;
}

static __device__ inline v8f wmma_bf16(v16bf a, v16bf b, v8f c) {
    return __builtin_amdgcn_wmma_f32_16x16x32_bf16(
        false, a, false, b, (short)0, c, false, false);
}

// A-operand (16x32 bf16), row-major source. Two 16B contiguous loads per lane.
static __device__ inline v16bf load_a(const unsigned short* A, int lda, int row0, int k0) {
    int lane = threadIdx.x & 31;
    int row  = row0 + (lane & 15);
    int kb   = (lane < 16) ? 0 : 8;
    const unsigned short* p = A + (size_t)row * lda + k0 + kb;
    V16 t;
    t.q[0] = *(const U4*)(p);
    t.q[1] = *(const U4*)(p + 16);
    return t.v;
}

static __device__ inline v16bf load_a_clamp(const unsigned short* A, int lda,
                                            int row0, int k0, int maxrow) {
    int lane = threadIdx.x & 31;
    int row  = row0 + (lane & 15);
    if (row > maxrow) row = maxrow;
    int kb   = (lane < 16) ? 0 : 8;
    const unsigned short* p = A + (size_t)row * lda + k0 + kb;
    V16 t;
    t.q[0] = *(const U4*)(p);
    t.q[1] = *(const U4*)(p + 16);
    return t.v;
}

// B-operand (32x16 bf16) where B[k][col] = M[row(col)][k]; `row` is the
// per-lane selected column's source row; K runs contiguously in memory.
// Two 16B contiguous loads per lane.
static __device__ inline v16bf load_b_rows(const unsigned short* M, int ld,
                                           int row, int dbase) {
    int lane = threadIdx.x & 31;
    int kb   = dbase + ((lane < 16) ? 0 : 8);
    const unsigned short* p = M + (size_t)row * ld + kb;
    V16 t;
    t.q[0] = *(const U4*)(p);
    t.q[1] = *(const U4*)(p + 16);
    return t.v;
}

// ---------------- kernel 1a: fp32 -> bf16 (no transpose) ----------------
__global__ void cvt_bf16_kernel(const float* __restrict__ in,
                                unsigned short* __restrict__ out, long long n) {
    long long i = (long long)blockIdx.x * blockDim.x + threadIdx.x;
    if (i < n) out[i] = f2bf(in[i]);
}

// ---------------- kernel 1b: fp32 [K][N] -> bf16 [N][K] (transpose) -------
__global__ void cvt_bf16_t_kernel(const float* __restrict__ in,
                                  unsigned short* __restrict__ out,
                                  int K, int N) {
    long long i = (long long)blockIdx.x * blockDim.x + threadIdx.x;
    if (i >= (long long)K * N) return;
    int k = (int)(i / N);
    int n = (int)(i % N);
    out[(size_t)n * K + k] = f2bf(in[i]);
}

// ---------------- kernel 2: positional embedding (bf16) ----------------
__global__ void pos_embed_kernel(unsigned short* __restrict__ pos) {
    long long idx = (long long)blockIdx.x * blockDim.x + threadIdx.x;
    if (idx >= (long long)LL * FF) return;
    int jp = (int)(idx / FF);
    int f  = (int)(idx % FF);
    int d  = jp - (NN - 1);
    const int nb = FF / 2;
    int fe = (f < nb) ? f : f - nb;
    float cw = __expf((float)(fe + 1) * (__logf((float)(NN + 1)) / (float)nb)) - 1.0f;
    float e  = (cw > fabsf((float)d)) ? 1.0f : 0.0f;
    float val = (f < nb) ? e : ((d > 0 ? 1.0f : (d < 0 ? -1.0f : 0.0f)) * e);
    pos[idx] = f2bf(val);
}

// ---------------- kernel 3: QKV projection GEMM (32x64 tile / wave) -------
// z=0: q -> qc (scale+content bias) + qr (scale+pos bias), layout [H][N][D]
// z=1: k -> [H][N][D]   z=2: v -> transposed [H][D][N] (packed 16B stores)
__global__ __launch_bounds__(32)
void gemm_qkv_kernel(const unsigned short* __restrict__ xb,
                     const unsigned short* __restrict__ wqt,
                     const unsigned short* __restrict__ wkt,
                     const unsigned short* __restrict__ wvt,
                     const float* __restrict__ cbias,
                     const float* __restrict__ pbias,
                     unsigned short* __restrict__ qc,
                     unsigned short* __restrict__ qr,
                     unsigned short* __restrict__ kout,
                     unsigned short* __restrict__ vT) {
    int m0 = blockIdx.x * 32;
    int n0 = blockIdx.y * 64;
    int which = blockIdx.z;
    const unsigned short* Wt = (which == 0) ? wqt : (which == 1) ? wkt : wvt;

    int lane = threadIdx.x & 31;
    int hw   = lane >> 4;
    int jl   = lane & 15;

    v8f acc[2][4];
#pragma unroll
    for (int mt = 0; mt < 2; ++mt)
#pragma unroll
        for (int nt = 0; nt < 4; ++nt) acc[mt][nt] = vzero8();

    for (int k0 = 0; k0 < DM; k0 += 32) {
        v16bf a0 = load_a(xb, DM, m0,      k0);
        v16bf a1 = load_a(xb, DM, m0 + 16, k0);
#pragma unroll
        for (int nt = 0; nt < 4; ++nt) {
            v16bf b = load_b_rows(Wt, DM, n0 + nt * 16 + jl, k0);
            acc[0][nt] = wmma_bf16(a0, b, acc[0][nt]);
            acc[1][nt] = wmma_bf16(a1, b, acc[1][nt]);
        }
    }

#pragma unroll
    for (int mt = 0; mt < 2; ++mt)
#pragma unroll
        for (int nt = 0; nt < 4; ++nt) {
            int col = n0 + nt * 16 + jl;
            int h   = col >> 6;
            int dd  = col & 63;
            int rowb = m0 + mt * 16 + 8 * hw;
            if (which == 2) {
                // transposed V: 8 consecutive rows are contiguous along N
                P8 pk;
#pragma unroll
                for (int r = 0; r < 8; ++r) pk.u[r] = f2bf(acc[mt][nt][r]);
                *(U4*)(vT + ((size_t)h * DD + dd) * NN + rowb) = pk.q;
            } else if (which == 1) {
#pragma unroll
                for (int r = 0; r < 8; ++r)
                    kout[((size_t)h * NN + rowb + r) * DD + dd] = f2bf(acc[mt][nt][r]);
            } else {
                float cb = cbias[h * DD + dd];
                float pb = pbias[h * DD + dd];
#pragma unroll
                for (int r = 0; r < 8; ++r) {
                    float q = acc[mt][nt][r] * QSCALE;
                    size_t oidx = ((size_t)h * NN + rowb + r) * DD + dd;
                    qc[oidx] = f2bf(q + cb);
                    qr[oidx] = f2bf(q + pb);
                }
            }
        }
}

// ---------------- kernel 4: rel_k = pos @ W_rel_k (32x64 tile) ------------
__global__ __launch_bounds__(32)
void gemm_relk_kernel(const unsigned short* __restrict__ pos,
                      const unsigned short* __restrict__ wrelt,
                      unsigned short* __restrict__ relk) {
    int m0 = blockIdx.x * 32;
    int n0 = blockIdx.y * 64;
    int lane = threadIdx.x & 31;
    int hw   = lane >> 4;
    int jl   = lane & 15;

    v8f acc[2][4];
#pragma unroll
    for (int mt = 0; mt < 2; ++mt)
#pragma unroll
        for (int nt = 0; nt < 4; ++nt) acc[mt][nt] = vzero8();

    for (int k0 = 0; k0 < FF; k0 += 32) {
        v16bf a0 = load_a_clamp(pos, FF, m0,      k0, LL - 1);
        v16bf a1 = load_a_clamp(pos, FF, m0 + 16, k0, LL - 1);
#pragma unroll
        for (int nt = 0; nt < 4; ++nt) {
            v16bf b = load_b_rows(wrelt, FF, n0 + nt * 16 + jl, k0);
            acc[0][nt] = wmma_bf16(a0, b, acc[0][nt]);
            acc[1][nt] = wmma_bf16(a1, b, acc[1][nt]);
        }
    }

#pragma unroll
    for (int mt = 0; mt < 2; ++mt)
#pragma unroll
        for (int nt = 0; nt < 4; ++nt) {
            int col = n0 + nt * 16 + jl;
            int h   = col >> 6;
            int dd  = col & 63;
#pragma unroll
            for (int r = 0; r < 8; ++r) {
                int jp = m0 + mt * 16 + r + 8 * hw;
                if (jp < LL)
                    relk[((size_t)h * LL + jp) * DD + dd] = f2bf(acc[mt][nt][r]);
            }
        }
}

// ---------------- kernel 5: flash attention with in-register rel-shift ----
__global__ __launch_bounds__(32)
void flash_attn_kernel(const unsigned short* __restrict__ qc,
                       const unsigned short* __restrict__ qr,
                       const unsigned short* __restrict__ kbuf,
                       const unsigned short* __restrict__ vT,
                       const unsigned short* __restrict__ relk,
                       unsigned short* __restrict__ attn_out) {
    __shared__ __align__(16) unsigned short sP[16 * 32];  // P tile (A-layout src)

    int lane = threadIdx.x & 31;
    int hw   = lane >> 4;
    int jl   = lane & 15;
    int i0   = blockIdx.x * 16;
    int h    = blockIdx.y;

    const unsigned short* qch = qc   + (size_t)h * NN * DD;
    const unsigned short* qrh = qr   + (size_t)h * NN * DD;
    const unsigned short* kh  = kbuf + (size_t)h * NN * DD;
    const unsigned short* vTh = vT   + (size_t)h * DD * NN;   // [64][N]
    const unsigned short* rh  = relk + (size_t)h * LL * DD;

    v16bf aq0 = load_a(qch, DD, i0, 0);
    v16bf aq1 = load_a(qch, DD, i0, 32);
    v16bf ar0 = load_a(qrh, DD, i0, 0);
    v16bf ar1 = load_a(qrh, DD, i0, 32);

    float mrun[8], lrun[8];
#pragma unroll
    for (int r = 0; r < 8; ++r) { mrun[r] = -3.0e38f; lrun[r] = 0.0f; }
    v8f o[4];
#pragma unroll
    for (int t = 0; t < 4; ++t) o[t] = vzero8();

    for (int it = 0; it < NN / 32; ++it) {
        int j0 = it * 32;
        if (j0 + 32 < NN) {
            __builtin_prefetch(kh + (size_t)(j0 + 32 + lane) * DD, 0, 1);
            __builtin_prefetch(vTh + (size_t)(lane << 1) * NN + j0 + 32, 0, 1);
        }
        v8f Ls[2];
#pragma unroll
        for (int s = 0; s < 2; ++s) {
            int jb = j0 + s * 16;
            // ---- content logits: Q x K^T ----
            v16bf bk0 = load_b_rows(kh, DD, jb + jl, 0);
            v16bf bk1 = load_b_rows(kh, DD, jb + jl, 32);
            v8f c = vzero8();
            c = wmma_bf16(aq0, bk0, c);
            c = wmma_bf16(aq1, bk1, c);
            // ---- rel logits band: two WMMA vs shifted rel_k rows ----
            int delta = jb - i0 + (NN - 1);
            int r1 = delta - 15 + jl;                       // >= 0 always
            int r2 = delta + 1 + jl; if (r2 > LL - 1) r2 = LL - 1;
            v16bf b10 = load_b_rows(rh, DD, r1, 0);
            v16bf b11 = load_b_rows(rh, DD, r1, 32);
            v16bf b20 = load_b_rows(rh, DD, r2, 0);
            v16bf b21 = load_b_rows(rh, DD, r2, 32);
            v8f p1 = vzero8(); p1 = wmma_bf16(ar0, b10, p1); p1 = wmma_bf16(ar1, b11, p1);
            v8f p2 = vzero8(); p2 = wmma_bf16(ar0, b20, p2); p2 = wmma_bf16(ar1, b21, p2);
            // ---- relative shift: cross-lane gather within each 16-lane half
            v8f lg;
#pragma unroll
            for (int r = 0; r < 8; ++r) {
                int il = r + 8 * hw;
                int m  = jl - il + 15;                      // 0..30
                int s1 = 16 * hw + (m < 16 ? m : 15);
                int s2 = 16 * hw + (m >= 16 ? m - 16 : 0);
                float v1 = __shfl(p1[r], s1, 32);
                float v2 = __shfl(p2[r], s2, 32);
                lg[r] = c[r] + ((m < 16) ? v1 : v2);
            }
            Ls[s] = lg;
        }
        // ---- online softmax over the 32-key pair ----
        float alpha[8];
#pragma unroll
        for (int r = 0; r < 8; ++r) {
            float rm = fmaxf(Ls[0][r], Ls[1][r]);
            rm = fmaxf(rm, __shfl_xor(rm, 1, 32));
            rm = fmaxf(rm, __shfl_xor(rm, 2, 32));
            rm = fmaxf(rm, __shfl_xor(rm, 4, 32));
            rm = fmaxf(rm, __shfl_xor(rm, 8, 32));
            float mn = fmaxf(mrun[r], rm);
            alpha[r] = __expf(mrun[r] - mn);
            float p0 = __expf(Ls[0][r] - mn);
            float p1 = __expf(Ls[1][r] - mn);
            float rs = p0 + p1;
            rs += __shfl_xor(rs, 1, 32);
            rs += __shfl_xor(rs, 2, 32);
            rs += __shfl_xor(rs, 4, 32);
            rs += __shfl_xor(rs, 8, 32);
            lrun[r] = lrun[r] * alpha[r] + rs;
            mrun[r] = mn;
            int il = r + 8 * hw;
            sP[il * 32 + jl]      = f2bf(p0);
            sP[il * 32 + 16 + jl] = f2bf(p1);
        }
#pragma unroll
        for (int t = 0; t < 4; ++t)
#pragma unroll
            for (int r = 0; r < 8; ++r) o[t][r] *= alpha[r];

        __syncthreads();
        // ---- O += P(16x32) x V(32x64), V read directly from vT[h][d][N] --
        v16bf pa = load_a(sP, 32, 0, 0);
#pragma unroll
        for (int t = 0; t < 4; ++t) {
            v16bf bv = load_b_rows(vTh, NN, t * 16 + jl, j0);
            o[t] = wmma_bf16(pa, bv, o[t]);
        }
        __syncthreads();
    }

    // ---- epilogue: normalize and store [N][HD] bf16 ----
#pragma unroll
    for (int t = 0; t < 4; ++t)
#pragma unroll
        for (int r = 0; r < 8; ++r) {
            int row = i0 + r + 8 * hw;
            float val = o[t][r] / lrun[r];
            attn_out[(size_t)row * HD + h * DD + t * 16 + jl] = f2bf(val);
        }
}

// ---------------- kernel 6: output projection + bias (fp32 out) ----------
__global__ __launch_bounds__(32)
void gemm_out_kernel(const unsigned short* __restrict__ attn,
                     const unsigned short* __restrict__ woutt,
                     const float* __restrict__ bout,
                     float* __restrict__ out) {
    int m0 = blockIdx.x * 32;
    int n0 = blockIdx.y * 64;
    int lane = threadIdx.x & 31;
    int hw   = lane >> 4;
    int jl   = lane & 15;

    v8f acc[2][4];
#pragma unroll
    for (int mt = 0; mt < 2; ++mt)
#pragma unroll
        for (int nt = 0; nt < 4; ++nt) acc[mt][nt] = vzero8();

    for (int k0 = 0; k0 < HD; k0 += 32) {
        v16bf a0 = load_a(attn, HD, m0,      k0);
        v16bf a1 = load_a(attn, HD, m0 + 16, k0);
#pragma unroll
        for (int nt = 0; nt < 4; ++nt) {
            v16bf b = load_b_rows(woutt, HD, n0 + nt * 16 + jl, k0);
            acc[0][nt] = wmma_bf16(a0, b, acc[0][nt]);
            acc[1][nt] = wmma_bf16(a1, b, acc[1][nt]);
        }
    }

#pragma unroll
    for (int mt = 0; mt < 2; ++mt)
#pragma unroll
        for (int nt = 0; nt < 4; ++nt) {
            int col = n0 + nt * 16 + jl;
            float bb = bout[col];
#pragma unroll
            for (int r = 0; r < 8; ++r) {
                int row = m0 + mt * 16 + r + 8 * hw;
                out[(size_t)row * DM + col] = acc[mt][nt][r] + bb;
            }
        }
}

// ---------------- workspace layout (ushort elements, 16-aligned) ----------
static inline size_t align16(size_t x) { return (x + 15) & ~(size_t)15; }

extern "C" void kernel_launch(void* const* d_in, const int* in_sizes, int n_in,
                              void* d_out, int out_size, void* d_ws, size_t ws_size,
                              hipStream_t stream) {
    const float* x     = (const float*)d_in[0];
    const float* W_q   = (const float*)d_in[1];
    const float* W_k   = (const float*)d_in[2];
    const float* W_v   = (const float*)d_in[3];
    const float* W_rel = (const float*)d_in[4];
    const float* W_out = (const float*)d_in[5];
    const float* b_out = (const float*)d_in[6];
    const float* cbias = (const float*)d_in[7];
    const float* pbias = (const float*)d_in[8];
    float* out = (float*)d_out;

    unsigned short* ws = (unsigned short*)d_ws;
    size_t off = 0;
    unsigned short* xb    = ws + off; off = align16(off + (size_t)NN * DM);
    unsigned short* wqt   = ws + off; off = align16(off + (size_t)DM * HD);
    unsigned short* wkt   = ws + off; off = align16(off + (size_t)DM * HD);
    unsigned short* wvt   = ws + off; off = align16(off + (size_t)DM * HD);
    unsigned short* wrelt = ws + off; off = align16(off + (size_t)FF * HD);
    unsigned short* woutt = ws + off; off = align16(off + (size_t)HD * DM);
    unsigned short* posb  = ws + off; off = align16(off + (size_t)LL * FF);
    unsigned short* qc    = ws + off; off = align16(off + (size_t)HH * NN * DD);
    unsigned short* qr    = ws + off; off = align16(off + (size_t)HH * NN * DD);
    unsigned short* kbuf  = ws + off; off = align16(off + (size_t)HH * NN * DD);
    unsigned short* vT    = ws + off; off = align16(off + (size_t)HH * NN * DD);
    unsigned short* relk  = ws + off; off = align16(off + (size_t)HH * LL * DD);
    unsigned short* attn  = ws + off; off = align16(off + (size_t)NN * HD);
    (void)ws_size; (void)in_sizes; (void)n_in; (void)out_size;

    // 1) converts: x plain; all weights transposed to [N][K]
    {
        long long n = (long long)NN * DM;
        cvt_bf16_kernel<<<(unsigned)((n + 255) / 256), 256, 0, stream>>>(x, xb, n);
    }
    auto cvtT = [&](const float* src, unsigned short* dst, int K, int N) {
        long long n = (long long)K * N;
        cvt_bf16_t_kernel<<<(unsigned)((n + 255) / 256), 256, 0, stream>>>(src, dst, K, N);
    };
    cvtT(W_q,   wqt,   DM, HD);
    cvtT(W_k,   wkt,   DM, HD);
    cvtT(W_v,   wvt,   DM, HD);
    cvtT(W_rel, wrelt, FF, HD);
    cvtT(W_out, woutt, HD, DM);

    // 2) positional embedding
    {
        long long n = (long long)LL * FF;
        pos_embed_kernel<<<(unsigned)((n + 255) / 256), 256, 0, stream>>>(posb);
    }

    // 3) QKV projections (fused scale/biases; V stored transposed)
    gemm_qkv_kernel<<<dim3(NN / 32, HD / 64, 3), 32, 0, stream>>>(
        xb, wqt, wkt, wvt, cbias, pbias, qc, qr, kbuf, vT);

    // 4) rel_k projection (8192 rows, last clamped/guarded)
    gemm_relk_kernel<<<dim3((LL + 31) / 32, HD / 64), 32, 0, stream>>>(
        posb, wrelt, relk);

    // 5) flash attention with in-register relative shift
    flash_attn_kernel<<<dim3(NN / 16, HH), 32, 0, stream>>>(
        qc, qr, kbuf, vT, relk, attn);

    // 6) output projection + bias
    gemm_out_kernel<<<dim3(NN / 32, DM / 64), 32, 0, stream>>>(
        attn, woutt, b_out, out);
}
